// Seq2Seq_39453569581639
// MI455X (gfx1250) — matrix operand order repaired and checked by reading, not compiled, not run on hardware
//
#include <hip/hip_runtime.h>

// ---------------- problem constants (match reference setup_inputs) ----------------
constexpr int B    = 64;
constexpr int S    = 1024;
constexpr int DIN  = 32;
constexpr int DOUT = 32;
constexpr int H    = 512;
constexpr int G4H  = 4 * H;        // 2048 gate rows (i,f,g,o)
constexpr int HB   = B * H;        // elements in one hidden state buffer
constexpr int NWG  = 8;            // cooperative workgroups (all co-resident)
constexpr int THREADS = 512;       // 16 waves/WG -> 128 waves = 32 j-tiles x 4 m-tiles

typedef __attribute__((ext_vector_type(16))) _Float16 v16h;
typedef __attribute__((ext_vector_type(8)))  float    v8f;
typedef __attribute__((ext_vector_type(4)))  unsigned u32x4;

union HQ { v16h h; u32x4 q[2]; };

// Compiler-level memory barrier: stops LICM from hoisting the (t-invariant)
// weight fragment loads out of the timestep loop, which previously caused a
// ~500-VGPR live range and massive scratch spills/reloads in the hot loop.
#define MEM_OPAQUE() asm volatile("" ::: "memory")

// ---- WMMA fragment loaders (CDNA5 16-bit operand layouts, wave32) ----
// A (16x32, M x K): lanes 0-15 row M=lane hold K {kb..kb+7, kb+16..kb+23};
//                   lanes 16-31 row M=lane-16 hold K {kb+8.., kb+24..}.
__device__ __forceinline__ v16h load_frag_a(const _Float16* __restrict__ base,
                                            int row0, int rowstride, int kbase, int lane) {
  const int lr = lane & 15;
  const int hf = lane >> 4;
  const _Float16* p = base + (size_t)(row0 + lr) * rowstride + kbase + hf * 8;
  HQ u;
  u.q[0] = *(const u32x4*)(p);
  u.q[1] = *(const u32x4*)(p + 16);
  return u.h;
}

// B (32x16, K x N): lane n (0-15) holds column n, K {kb..kb+7, kb+16..kb+23};
//                   lanes 16-31 hold K {kb+8.., kb+24..}. W stored row-major [N,K],
//                   so a B column == a contiguous row of W.
__device__ __forceinline__ v16h load_frag_b(const _Float16* __restrict__ w,
                                            int n0, int rowstride, int kbase, int lane) {
  const int lr = lane & 15;
  const int hf = lane >> 4;
  const _Float16* p = w + (size_t)(n0 + lr) * rowstride + kbase + hf * 8;
  HQ u;
  u.q[0] = *(const u32x4*)(p);
  u.q[1] = *(const u32x4*)(p + 16);
  return u.h;
}

__device__ __forceinline__ float sigm(float x) { return 1.0f / (1.0f + __expf(-x)); }
__device__ __forceinline__ float tanh_fast(float x) {
  x = fminf(15.0f, fmaxf(-15.0f, x));
  float e = __expf(-2.0f * x);
  return (1.0f - e) / (1.0f + e);
}

// Accumulate the 4 gate tiles (i,f,g,o) of output tile (m, j):
// gates[:, g*512 + 16j .. +16) += x[m-tile, :] @ W[g*512+16j .. , :]^T
__device__ __forceinline__ void gemm4(v8f acc[4],
                                      const _Float16* __restrict__ xin, int xstride, int xcol,
                                      const _Float16* __restrict__ W, int wstride, int ksteps,
                                      int j, int m, int lane) {
  MEM_OPAQUE();   // pin all fragment loads inside this GEMM / this timestep
#pragma unroll 4
  for (int kc = 0; kc < ksteps; ++kc) {
    v16h a = load_frag_a(xin + xcol, m * 16, xstride, kc * 32, lane);
#pragma unroll
    for (int g = 0; g < 4; ++g) {
      v16h b = load_frag_b(W, g * H + j * 16, wstride, kc * 32, lane);
      acc[g] = __builtin_amdgcn_wmma_f32_16x16x32_f16(false, a, false, b,
                                                      (short)0, acc[g], false, false);
    }
  }
}

// LSTM pointwise: c = sig(f)*c + sig(i)*tanh(g); h = sig(o)*tanh(c).
// C/D layout: VGPR e holds row (16m + e + 8*(lane>=16)), column 16j + lane%16.
__device__ __forceinline__ void cell_update(v8f acc[4], const float* __restrict__ bias,
                                            int j, int m, int lane, v8f& c,
                                            _Float16* __restrict__ hout) {
  const int lr = lane & 15;
  const int hf = lane >> 4;
  const int col = j * 16 + lr;
  const float bi = bias[0 * H + col];
  const float bf = bias[1 * H + col];
  const float bg = bias[2 * H + col];
  const float bo = bias[3 * H + col];
#pragma unroll
  for (int e = 0; e < 8; ++e) {
    float iv = sigm(acc[0][e] + bi);
    float fv = sigm(acc[1][e] + bf);
    float gv = tanh_fast(acc[2][e] + bg);
    float ov = sigm(acc[3][e] + bo);
    float cv = fv * c[e] + iv * gv;
    c[e] = cv;
    float hv = ov * tanh_fast(cv);
    int row = m * 16 + e + hf * 8;
    hout[row * H + col] = (_Float16)hv;
  }
}

// Device-scope barrier across the 8 persistent workgroups (monotonic counter,
// re-zeroed by an init kernel on every launch -> deterministic replays).
__device__ __forceinline__ void grid_barrier(unsigned* bar, unsigned& epoch) {
  epoch += 1;
  const unsigned target = epoch * NWG;
  __threadfence();
  __syncthreads();
  if (threadIdx.x == 0) {
    atomicAdd(bar, 1u);
    while (*((volatile unsigned*)bar) < target) __builtin_amdgcn_s_sleep(2);
  }
  __syncthreads();
  __threadfence();
}

__global__ __launch_bounds__(THREADS)
void seq2seq_main(const _Float16* __restrict__ src16,
                  const _Float16* __restrict__ eWih0, const _Float16* __restrict__ eWhh0,
                  const _Float16* __restrict__ eWih1, const _Float16* __restrict__ eWhh1,
                  const _Float16* __restrict__ dWih0, const _Float16* __restrict__ dWhh0,
                  const _Float16* __restrict__ dWih1, const _Float16* __restrict__ dWhh1,
                  const _Float16* __restrict__ outW,
                  const float* __restrict__ eb0, const float* __restrict__ eb1,
                  const float* __restrict__ db0, const float* __restrict__ db1,
                  const float* __restrict__ outb,
                  _Float16* __restrict__ h0, _Float16* __restrict__ h1,
                  _Float16* __restrict__ xbuf,
                  unsigned* __restrict__ bar,
                  float* __restrict__ out,
                  const int* __restrict__ tlen) {
  const int lane = threadIdx.x & 31;
  const int wave = threadIdx.x >> 5;
  const int gw   = blockIdx.x * (THREADS / 32) + wave;   // 0..127
  const int j = gw >> 2;       // 0..31 : hidden-column tile (owns c for these cols)
  const int m = gw & 3;        // 0..3  : batch-row tile
  const int T = *tlen;

  v8f c0 = {};                 // layer-0 cell state, resident in VGPRs whole run
  v8f c1 = {};                 // layer-1 cell state
  unsigned epoch = 0;
  int p = 0;                   // ping-pong index for h buffers

  // ---------------- encoder: 1024 steps ----------------
  for (int t = 0; t < S; ++t) {
    {
      v8f acc[4] = {};
      gemm4(acc, src16, S * DIN, t * DIN, eWih0, DIN, 1, j, m, lane);       // K=32
      gemm4(acc, h0 + p * HB, H, 0, eWhh0, H, H / 32, j, m, lane);          // K=512
      cell_update(acc, eb0, j, m, lane, c0, h0 + (1 - p) * HB);
    }
    grid_barrier(bar, epoch);
    {
      v8f acc[4] = {};
      gemm4(acc, h0 + (1 - p) * HB, H, 0, eWih1, H, H / 32, j, m, lane);    // K=512
      gemm4(acc, h1 + p * HB, H, 0, eWhh1, H, H / 32, j, m, lane);          // K=512
      cell_update(acc, eb1, j, m, lane, c1, h1 + (1 - p) * HB);
    }
    grid_barrier(bar, epoch);
    p ^= 1;
  }

  // ---------------- decoder: T autoregressive steps ----------------
  for (int t = 0; t < T; ++t) {
    {
      v8f acc[4] = {};
      gemm4(acc, xbuf, DOUT, 0, dWih0, DOUT, 1, j, m, lane);                // K=32
      gemm4(acc, h0 + p * HB, H, 0, dWhh0, H, H / 32, j, m, lane);
      cell_update(acc, db0, j, m, lane, c0, h0 + (1 - p) * HB);
    }
    grid_barrier(bar, epoch);
    {
      v8f acc[4] = {};
      gemm4(acc, h0 + (1 - p) * HB, H, 0, dWih1, H, H / 32, j, m, lane);
      gemm4(acc, h1 + p * HB, H, 0, dWhh1, H, H / 32, j, m, lane);
      cell_update(acc, db1, j, m, lane, c1, h1 + (1 - p) * HB);
    }
    grid_barrier(bar, epoch);
    // output projection pred = h1 @ outW^T + outb : M=64, N=32, K=512 -> 8 tiles
    if (gw < 8) {
      const int n2 = gw >> 2;      // 0..1 output column tile
      const int pm = gw & 3;       // 0..3 batch row tile
      v8f acc = {};
      MEM_OPAQUE();
#pragma unroll 4
      for (int kc = 0; kc < H / 32; ++kc) {
        v16h a = load_frag_a(h1 + (1 - p) * HB, pm * 16, H, kc * 32, lane);
        v16h b = load_frag_b(outW, n2 * 16, H, kc * 32, lane);
        acc = __builtin_amdgcn_wmma_f32_16x16x32_f16(false, a, false, b,
                                                     (short)0, acc, false, false);
      }
      const int lr = lane & 15, hf = lane >> 4;
      const int col = n2 * 16 + lr;
      const float bv = outb[col];
#pragma unroll
      for (int e = 0; e < 8; ++e) {
        float v = acc[e] + bv;
        int row = pm * 16 + e + hf * 8;
        out[(size_t)row * T * DOUT + (size_t)t * DOUT + col] = v;   // fp32 output
        xbuf[row * DOUT + col] = (_Float16)v;                       // feedback input
      }
    }
    grid_barrier(bar, epoch);
    p ^= 1;
  }
}

// ---------------- prep kernels (run every launch; capture-safe) ----------------
__global__ void k_f32_to_f16(const float* __restrict__ a, _Float16* __restrict__ o, int n) {
  int i = blockIdx.x * blockDim.x + threadIdx.x;
  if (i < n) o[i] = (_Float16)a[i];
}
__global__ void k_bias_sum(const float* __restrict__ a, const float* __restrict__ b,
                           float* __restrict__ o, int n) {
  int i = blockIdx.x * blockDim.x + threadIdx.x;
  if (i < n) o[i] = a[i] + b[i];
}
__global__ void k_zero_f16(_Float16* __restrict__ p, int n) {
  int i = blockIdx.x * blockDim.x + threadIdx.x;
  if (i < n) p[i] = (_Float16)0.0f;
}
__global__ void k_zero_u32(unsigned* __restrict__ p, int n) {
  int i = blockIdx.x * blockDim.x + threadIdx.x;
  if (i < n) p[i] = 0u;
}

extern "C" void kernel_launch(void* const* d_in, const int* in_sizes, int n_in,
                              void* d_out, int out_size, void* d_ws, size_t ws_size,
                              hipStream_t stream) {
  (void)in_sizes; (void)n_in; (void)out_size; (void)ws_size;
  const float* src    = (const float*)d_in[0];
  const int*   tlen   = (const int*)d_in[1];
  const float* eWih0f = (const float*)d_in[2];
  const float* eWhh0f = (const float*)d_in[3];
  const float* ebih0  = (const float*)d_in[4];
  const float* ebhh0  = (const float*)d_in[5];
  const float* eWih1f = (const float*)d_in[6];
  const float* eWhh1f = (const float*)d_in[7];
  const float* ebih1  = (const float*)d_in[8];
  const float* ebhh1  = (const float*)d_in[9];
  const float* dWih0f = (const float*)d_in[10];
  const float* dWhh0f = (const float*)d_in[11];
  const float* dbih0  = (const float*)d_in[12];
  const float* dbhh0  = (const float*)d_in[13];
  const float* dWih1f = (const float*)d_in[14];
  const float* dWhh1f = (const float*)d_in[15];
  const float* dbih1  = (const float*)d_in[16];
  const float* dbhh1  = (const float*)d_in[17];
  const float* outWf  = (const float*)d_in[18];
  const float* outbf  = (const float*)d_in[19];
  float* outp = (float*)d_out;

  // workspace layout (256B aligned regions), ~17.5 MB total
  char* ws = (char*)d_ws;
  size_t off = 0;
  auto take = [&](size_t bytes) -> char* {
    char* p = ws + off;
    off = (off + bytes + 255) & ~(size_t)255;
    return p;
  };
  unsigned*  bar   = (unsigned*) take(4);
  _Float16*  xbuf  = (_Float16*) take((size_t)B * DOUT * 2);
  _Float16*  h0    = (_Float16*) take((size_t)2 * HB * 2);
  _Float16*  h1    = (_Float16*) take((size_t)2 * HB * 2);
  _Float16*  src16 = (_Float16*) take((size_t)B * S * DIN * 2);
  _Float16*  eWih0 = (_Float16*) take((size_t)G4H * DIN * 2);
  _Float16*  eWhh0 = (_Float16*) take((size_t)G4H * H * 2);
  _Float16*  eWih1 = (_Float16*) take((size_t)G4H * H * 2);
  _Float16*  eWhh1 = (_Float16*) take((size_t)G4H * H * 2);
  _Float16*  dWih0 = (_Float16*) take((size_t)G4H * DOUT * 2);
  _Float16*  dWhh0 = (_Float16*) take((size_t)G4H * H * 2);
  _Float16*  dWih1 = (_Float16*) take((size_t)G4H * H * 2);
  _Float16*  dWhh1 = (_Float16*) take((size_t)G4H * H * 2);
  _Float16*  outW  = (_Float16*) take((size_t)DOUT * H * 2);
  float*     eb0   = (float*)    take((size_t)G4H * 4);
  float*     eb1   = (float*)    take((size_t)G4H * 4);
  float*     db0   = (float*)    take((size_t)G4H * 4);
  float*     db1   = (float*)    take((size_t)G4H * 4);

  auto cvt = [&](const float* a, _Float16* o, int n) {
    k_f32_to_f16<<<(n + 255) / 256, 256, 0, stream>>>(a, o, n);
  };
  cvt(src,    src16, B * S * DIN);
  cvt(eWih0f, eWih0, G4H * DIN);
  cvt(eWhh0f, eWhh0, G4H * H);
  cvt(eWih1f, eWih1, G4H * H);
  cvt(eWhh1f, eWhh1, G4H * H);
  cvt(dWih0f, dWih0, G4H * DOUT);
  cvt(dWhh0f, dWhh0, G4H * H);
  cvt(dWih1f, dWih1, G4H * H);
  cvt(dWhh1f, dWhh1, G4H * H);
  cvt(outWf,  outW,  DOUT * H);

  auto bsum = [&](const float* a, const float* b, float* o) {
    k_bias_sum<<<(G4H + 255) / 256, 256, 0, stream>>>(a, b, o, G4H);
  };
  bsum(ebih0, ebhh0, eb0);
  bsum(ebih1, ebhh1, eb1);
  bsum(dbih0, dbhh0, db0);
  bsum(dbih1, dbhh1, db1);

  k_zero_f16<<<(2 * HB + 255) / 256, 256, 0, stream>>>(h0, 2 * HB);
  k_zero_f16<<<(2 * HB + 255) / 256, 256, 0, stream>>>(h1, 2 * HB);
  k_zero_f16<<<(B * DOUT + 255) / 256, 256, 0, stream>>>(xbuf, B * DOUT);
  k_zero_u32<<<1, 32, 0, stream>>>(bar, 1);

  seq2seq_main<<<NWG, THREADS, 0, stream>>>(
      src16, eWih0, eWhh0, eWih1, eWhh1, dWih0, dWhh0, dWih1, dWhh1, outW,
      eb0, eb1, db0, db1, outbf, h0, h1, xbuf, bar, outp, tlen);
}
